// KoopmanAE_17111149707502
// MI455X (gfx1250) — matrix operand order repaired and checked by reading, not compiled, not run on hardware
//
#include <hip/hip_runtime.h>

// ---------------------------------------------------------------------------
// Types for CDNA5 WMMA (wave32)
// ---------------------------------------------------------------------------
typedef __attribute__((ext_vector_type(16))) __bf16 v16bf;
typedef __attribute__((ext_vector_type(8)))  float  v8f;

union FragBF {
    v16bf v;
    uint4 q[2];
};

__device__ __forceinline__ unsigned short f2bf(float f) {
    unsigned int u = __float_as_uint(f);
    u += 0x7FFFu + ((u >> 16) & 1u);   // round-to-nearest-even
    return (unsigned short)(u >> 16);
}

// CDNA5 async global->LDS copy (tracked by ASYNCcnt, no VGPR data path).
// vdst = per-lane LDS byte address, vaddr = per-lane 64-bit global address.
__device__ __forceinline__ void async_ld_b128(unsigned lds_addr, const void* g) {
    asm volatile("global_load_async_to_lds_b128 %0, %1, off"
                 :: "v"(lds_addr), "v"(g)
                 : "memory");
}
__device__ __forceinline__ void wait_async0() {
    asm volatile("s_wait_asynccnt 0x0" ::: "memory");
}

// ---------------------------------------------------------------------------
// fp32 -> bf16 conversion (grid-stride)
// ---------------------------------------------------------------------------
__global__ void cvt_f32_bf16(const float* __restrict__ s,
                             unsigned short* __restrict__ d, int n) {
    for (int i = blockIdx.x * blockDim.x + threadIdx.x; i < n;
         i += gridDim.x * blockDim.x)
        d[i] = f2bf(s[i]);
}

// ---------------------------------------------------------------------------
// Tustin discretization: Gauss-Jordan on [M | N | delta*B], M = I - d/2 A,
// N = I + d/2 A (128x288 augmented system in global ws, single workgroup).
// Writes bf16 Kd (128x128) and Ld (128x32).
// ---------------------------------------------------------------------------
__global__ __launch_bounds__(256)
void koopman_solve(const float* __restrict__ A, const float* __restrict__ B,
                   const float* __restrict__ delta_log,
                   float* __restrict__ G,
                   unsigned short* __restrict__ Kdb,
                   unsigned short* __restrict__ Ldb) {
    const int L = 128, C = 288;
    const int tid = threadIdx.x;
    __shared__ float colk[128];

    const float delta = expf(delta_log[0]);

    for (int idx = tid; idx < L * C; idx += 256) {
        int i = idx / C, j = idx % C;
        float v;
        if (j < 128) {
            v = (i == j ? 1.0f : 0.0f) - 0.5f * delta * A[i * 128 + j];
        } else if (j < 256) {
            int jj = j - 128;
            v = (i == jj ? 1.0f : 0.0f) + 0.5f * delta * A[i * 128 + jj];
        } else {
            v = delta * B[i * 32 + (j - 256)];
        }
        G[idx] = v;
    }
    __threadfence_block();
    __syncthreads();

    for (int k = 0; k < L; ++k) {
        for (int i = tid; i < L; i += 256) colk[i] = G[i * C + k];
        __threadfence_block();
        __syncthreads();
        float pivinv = 1.0f / colk[k];
        for (int j = tid; j < C; j += 256) G[k * C + j] *= pivinv;
        __threadfence_block();
        __syncthreads();
        for (int i = 0; i < L; ++i) {
            if (i == k) continue;
            float f = colk[i];
            for (int j = tid; j < C; j += 256)
                G[i * C + j] -= f * G[k * C + j];
        }
        __threadfence_block();
        __syncthreads();
    }

    for (int idx = tid; idx < 128 * 128; idx += 256) {
        int i = idx >> 7, j = idx & 127;
        Kdb[idx] = f2bf(G[i * C + 128 + j]);
    }
    for (int idx = tid; idx < 128 * 32; idx += 256) {
        int i = idx / 32, j = idx % 32;
        Ldb[idx] = f2bf(G[i * C + 256 + j]);
    }
}

// ---------------------------------------------------------------------------
// Tiled bf16 WMMA GEMM:  Y[N x M] = X[N x K] * W[M x K]^T (+bias)(+addend)
// Block: 256 threads = 8 waves. Block tile: 256 rows x (CTILES*16) cols.
// Each wave: 32 rows x CTILES*16 cols = 2 A-frags x CTILES B-tiles
//            -> 2*CTILES WMMA per K-step.
// X tile double-buffered in LDS (2 x 256x32 bf16 = 32 KB), filled by the
// CDNA5 async global->LDS engine (GLOBAL_LOAD_ASYNC_TO_LDS_B128): no staging
// VGPRs, no spill; the WMMA block hides the async-copy latency. Per-counter
// sync: s_wait_asynccnt 0 (own writes) then barrier (all waves) each step.
// W read from global bf16 (weights are L2/WGP$-resident).
// ---------------------------------------------------------------------------
template <int CTILES>
__global__ __launch_bounds__(256, 1)
void gemm_bf16_wmma(const unsigned short* __restrict__ Xb, int ldx,
                    const unsigned short* __restrict__ Wb,
                    const float* __restrict__ bias,
                    const float* __restrict__ addend, int lda,
                    float* __restrict__ outF, int ldf,
                    unsigned short* __restrict__ outB, int ldb,
                    int K, int relu) {
    __shared__ __align__(16) unsigned short ldsX[2][256 * 32];

    const int tid  = threadIdx.x;
    const int lane = tid & 31;
    const int wave = tid >> 5;          // 0..7 -> 32-row subtile
    const int half = (lane >> 4) & 1;   // lane 0-15 vs 16-31
    const int l16  = lane & 15;
    const long long rowBase = (long long)blockIdx.x * 256;
    const int colBase = blockIdx.y * (CTILES * 16);

    // per-thread staging chunk coordinates (4 chunks of 8 bf16 per thread)
    int srow[4], ssub[4];
#pragma unroll
    for (int h = 0; h < 4; ++h) {
        int c = tid + h * 256;
        srow[h] = c >> 2;
        ssub[h] = c & 3;
    }

    v8f acc0[CTILES] = {};
    v8f acc1[CTILES] = {};

    const int ksteps = K >> 5;

    // preload tile 0 via async engine
#pragma unroll
    for (int h = 0; h < 4; ++h)
        async_ld_b128(
            (unsigned)(uintptr_t)&ldsX[0][srow[h] * 32 + ssub[h] * 8],
            Xb + (rowBase + srow[h]) * ldx + ssub[h] * 8);

    for (int kt = 0; kt < ksteps; ++kt) {
        const int cur = kt & 1;
        wait_async0();                  // own async writes into lds[cur] done
        __syncthreads();                // all waves done -> lds[cur] published

        // kick off async fill of the other buffer (readers drained above)
        if (kt + 1 < ksteps) {
#pragma unroll
            for (int h = 0; h < 4; ++h)
                async_ld_b128(
                    (unsigned)(uintptr_t)
                        &ldsX[cur ^ 1][srow[h] * 32 + ssub[h] * 8],
                    Xb + (rowBase + srow[h]) * ldx + ((kt + 1) << 5) +
                        ssub[h] * 8);
        }

        // B fragments (shared by both A-frags): 16-bit B 32x16 layout,
        // lanes 0-15 hold K0..7,K16..23 of column l16, lanes 16-31 hold K8..15,K24..31
        FragBF b[CTILES];
#pragma unroll
        for (int j = 0; j < CTILES; ++j) {
            const unsigned short* wp =
                Wb + (long long)(colBase + j * 16 + l16) * K + (kt << 5) +
                half * 8;
            b[j].q[0] = *reinterpret_cast<const uint4*>(wp);
            b[j].q[1] = *reinterpret_cast<const uint4*>(wp + 16);
        }

        // A fragments: 16-bit A 16x32 layout
        //   lanes 0-15:  VGPR0-3 = K0..7,   VGPR4-7 = K16..23
        //   lanes 16-31: VGPR0-3 = K8..15,  VGPR4-7 = K24..31
        FragBF a0, a1;
        const int r0 = (wave * 32 + l16) * 32 + half * 8;
        const int r1 = (wave * 32 + 16 + l16) * 32 + half * 8;
        a0.q[0] = *reinterpret_cast<const uint4*>(&ldsX[cur][r0]);
        a0.q[1] = *reinterpret_cast<const uint4*>(&ldsX[cur][r0 + 16]);
        a1.q[0] = *reinterpret_cast<const uint4*>(&ldsX[cur][r1]);
        a1.q[1] = *reinterpret_cast<const uint4*>(&ldsX[cur][r1 + 16]);

#pragma unroll
        for (int j = 0; j < CTILES; ++j)
            acc0[j] = __builtin_amdgcn_wmma_f32_16x16x32_bf16(
                false, a0.v, false, b[j].v, (short)0, acc0[j], false, false);
#pragma unroll
        for (int j = 0; j < CTILES; ++j)
            acc1[j] = __builtin_amdgcn_wmma_f32_16x16x32_bf16(
                false, a1.v, false, b[j].v, (short)0, acc1[j], false, false);
    }

    // C/D layout: VGPR r -> M = r (lanes 0-15) / r+8 (lanes 16-31), N = l16
#pragma unroll
    for (int g = 0; g < 2; ++g) {
#pragma unroll
        for (int j = 0; j < CTILES; ++j) {
            int col = colBase + j * 16 + l16;
            float bv = bias ? bias[col] : 0.0f;
#pragma unroll
            for (int r = 0; r < 8; ++r) {
                long long row = rowBase + wave * 32 + g * 16 + half * 8 + r;
                float v = (g ? acc1[j][r] : acc0[j][r]) + bv;
                if (addend) v += addend[row * lda + col];
                if (relu)   v = fmaxf(v, 0.0f);
                if (outF) outF[row * ldf + col] = v;
                if (outB) outB[row * ldb + col] = f2bf(v);
            }
        }
    }
}

// ---------------------------------------------------------------------------
// Copy (64,512,W) -> (64,511,W): drop last timestep of each batch
// ---------------------------------------------------------------------------
__global__ void slice_copy(const float* __restrict__ src,
                           float* __restrict__ dst, int width) {
    long long total = 64LL * 511 * width;
    for (long long idx = (long long)blockIdx.x * blockDim.x + threadIdx.x;
         idx < total; idx += (long long)gridDim.x * blockDim.x) {
        int w = (int)(idx % width);
        long long row = idx / width;
        int b = (int)(row / 511), t = (int)(row % 511);
        dst[idx] = src[((long long)b * 512 + t) * width + w];
    }
}

// ---------------------------------------------------------------------------
// Host launcher
// ---------------------------------------------------------------------------
extern "C" void kernel_launch(void* const* d_in, const int* in_sizes, int n_in,
                              void* d_out, int out_size, void* d_ws, size_t ws_size,
                              hipStream_t stream) {
    (void)in_sizes; (void)n_in; (void)out_size; (void)ws_size;
    const float* x   = (const float*)d_in[0];
    const float* u   = (const float*)d_in[1];
    const float* eW1 = (const float*)d_in[2];
    const float* eb1 = (const float*)d_in[3];
    const float* eW2 = (const float*)d_in[4];
    const float* eb2 = (const float*)d_in[5];
    const float* eW3 = (const float*)d_in[6];
    const float* eb3 = (const float*)d_in[7];
    const float* dW1 = (const float*)d_in[8];
    const float* db1 = (const float*)d_in[9];
    const float* dW2 = (const float*)d_in[10];
    const float* db2 = (const float*)d_in[11];
    const float* dW3 = (const float*)d_in[12];
    const float* db3 = (const float*)d_in[13];
    const float* Am  = (const float*)d_in[14];
    const float* Bm  = (const float*)d_in[15];
    const float* dlg = (const float*)d_in[16];
    const float* aW  = (const float*)d_in[17];
    const float* ab  = (const float*)d_in[18];

    const int NT = 64 * 512;   // 32768 tokens

    char* base = (char*)d_ws;
    size_t off = 0;
    auto alloc = [&](size_t nbytes) -> void* {
        void* p = base + off;
        off += (nbytes + 255) & ~(size_t)255;
        return p;
    };

    // activations (bf16)
    unsigned short* xb    = (unsigned short*)alloc((size_t)NT * 256 * 2);
    unsigned short* ub    = (unsigned short*)alloc((size_t)NT * 32 * 2);
    unsigned short* uencb = (unsigned short*)alloc((size_t)NT * 32 * 2);
    unsigned short* encb  = (unsigned short*)alloc((size_t)NT * 128 * 2);
    unsigned short* h1b   = (unsigned short*)alloc((size_t)NT * 512 * 2);
    unsigned short* h2b   = (unsigned short*)alloc((size_t)NT * 512 * 2);
    unsigned short* plb   = (unsigned short*)alloc((size_t)NT * 128 * 2);
    // fp32 intermediates
    float* uLf   = (float*)alloc((size_t)NT * 128 * 4);
    float* plf   = (float*)alloc((size_t)NT * 128 * 4);
    float* predf = (float*)alloc((size_t)NT * 256 * 4);
    float* G     = (float*)alloc((size_t)128 * 288 * 4);
    // bf16 weights
    unsigned short* wE1  = (unsigned short*)alloc(512 * 256 * 2);
    unsigned short* wE2  = (unsigned short*)alloc(512 * 512 * 2);
    unsigned short* wE3  = (unsigned short*)alloc(128 * 512 * 2);
    unsigned short* wD1  = (unsigned short*)alloc(512 * 128 * 2);
    unsigned short* wD2  = (unsigned short*)alloc(512 * 512 * 2);
    unsigned short* wD3  = (unsigned short*)alloc(256 * 512 * 2);
    unsigned short* wAct = (unsigned short*)alloc(32 * 32 * 2);
    unsigned short* Kdb  = (unsigned short*)alloc(128 * 128 * 2);
    unsigned short* Ldb  = (unsigned short*)alloc(128 * 32 * 2);

    auto cvt = [&](const float* s, unsigned short* d, int n) {
        int blocks = (n + 255) / 256;
        if (blocks > 4096) blocks = 4096;
        cvt_f32_bf16<<<blocks, 256, 0, stream>>>(s, d, n);
    };
    cvt(x, xb, NT * 256);
    cvt(u, ub, NT * 32);
    cvt(eW1, wE1, 512 * 256);
    cvt(eW2, wE2, 512 * 512);
    cvt(eW3, wE3, 128 * 512);
    cvt(dW1, wD1, 512 * 128);
    cvt(dW2, wD2, 512 * 512);
    cvt(dW3, wD3, 256 * 512);
    cvt(aW,  wAct, 32 * 32);

    koopman_solve<<<1, 256, 0, stream>>>(Am, Bm, dlg, G, Kdb, Ldb);

    // M multiple of 64 -> CTILES=4; M==32 -> CTILES=2 (single column block)
    auto gemm = [&](const unsigned short* X, int K, const unsigned short* W, int M,
                    const float* bias, const float* add, int lda,
                    float* oF, int ldf, unsigned short* oB, int ldb, int relu) {
        if (M % 64 == 0) {
            dim3 grid(NT / 256, M / 64);
            gemm_bf16_wmma<4><<<grid, 256, 0, stream>>>(
                X, K, W, bias, add, lda, oF, ldf, oB, ldb, K, relu);
        } else {
            dim3 grid(NT / 256, 1);
            gemm_bf16_wmma<2><<<grid, 256, 0, stream>>>(
                X, K, W, bias, add, lda, oF, ldf, oB, ldb, K, relu);
        }
    };

    float* out_enc = (float*)d_out;                              // (64,512,128)
    float* out_rec = out_enc + (size_t)NT * 128;                 // (64,512,256)
    float* out_pl  = out_rec + (size_t)NT * 256;                 // (64,511,128)
    float* out_pr  = out_pl + (size_t)64 * 511 * 128;            // (64,511,256)

    // encoder
    gemm(xb, 256, wE1, 512, eb1, nullptr, 0, nullptr, 0, h1b, 512, 1);
    gemm(h1b, 512, wE2, 512, eb2, nullptr, 0, nullptr, 0, h2b, 512, 1);
    gemm(h2b, 512, wE3, 128, eb3, nullptr, 0, out_enc, 128, encb, 128, 0);
    // decoder (reconstructions)
    gemm(encb, 128, wD1, 512, db1, nullptr, 0, nullptr, 0, h1b, 512, 1);
    gemm(h1b, 512, wD2, 512, db2, nullptr, 0, nullptr, 0, h2b, 512, 1);
    gemm(h2b, 512, wD3, 256, db3, nullptr, 0, out_rec, 256, nullptr, 0, 0);
    // Koopman single-step latents
    gemm(ub, 32, wAct, 32, ab, nullptr, 0, nullptr, 0, uencb, 32, 0);
    gemm(uencb, 32, Ldb, 128, nullptr, nullptr, 0, uLf, 128, nullptr, 0, 0);
    gemm(encb, 128, Kdb, 128, nullptr, uLf, 128, plf, 128, plb, 128, 0);
    slice_copy<<<2048, 256, 0, stream>>>(plf, out_pl, 128);
    // decoder (predictions)
    gemm(plb, 128, wD1, 512, db1, nullptr, 0, nullptr, 0, h1b, 512, 1);
    gemm(h1b, 512, wD2, 512, db2, nullptr, 0, nullptr, 0, h2b, 512, 1);
    gemm(h2b, 512, wD3, 256, db3, nullptr, 0, predf, 256, nullptr, 0, 0);
    slice_copy<<<2048, 256, 0, stream>>>(predf, out_pr, 256);
}
